// MultiHeadSelfAttention_65120294142185
// MI455X (gfx1250) — matrix operand order repaired and checked by reading, not compiled, not run on hardware
//
#include <hip/hip_runtime.h>

#define D_MODEL   1024
#define NUM_HEADS 16
#define D_K       64
#define SEQ       2048
#define BATCH     2
#define ROWS      (BATCH * SEQ)     // 4096
#define QKV_N     (3 * D_MODEL)     // 3072

typedef __attribute__((ext_vector_type(16))) _Float16     v16h;
typedef __attribute__((ext_vector_type(8)))  _Float16     v8h;
typedef __attribute__((ext_vector_type(8)))  float        v8f;
typedef __attribute__((ext_vector_type(4)))  unsigned int u32x4;
typedef __attribute__((ext_vector_type(8)))  int          i32x8;
typedef __attribute__((ext_vector_type(4)))  int          i32x4;

// ---------------------------------------------------------------------------
// WMMA fragment helpers (CDNA5 wave32 layouts, CDNA5 ISA §7.12.2)
// All loads are 16-byte vector loads -> global_load_b128 / ds_load_b128.
// ---------------------------------------------------------------------------

__device__ __forceinline__ v16h concat8(v8h lo, v8h hi) {
  return __builtin_shufflevector(lo, hi, 0, 1, 2, 3, 4, 5, 6, 7, 8, 9, 10, 11,
                                 12, 13, 14, 15);
}

// A fragment: 16x32 f16 from row-major source (lda in elements).
// lanes 0-15: M=lane, K = 0..7 (elems 0..7) and 16..23 (elems 8..15)
// lanes 16-31: M=lane-16, K = 8..15 and 24..31
__device__ __forceinline__ v16h load_a16(const _Float16* __restrict__ src,
                                         int lda, int lane) {
  const int row = lane & 15;
  const int kb  = (lane & 16) ? 8 : 0;
  const _Float16* p = src + (size_t)row * lda + kb;
  const v8h lo = *(const v8h*)p;          // K kb..kb+7
  const v8h hi = *(const v8h*)(p + 16);   // K kb+16..kb+23
  return concat8(lo, hi);
}

// B fragment 32x16 where logical B(k,n) = W[n][k] (W row-major [N][K]):
// lanes 0-15: K = 0..7 and 16..23; lanes 16-31: K = 8..15 and 24..31 -> two b128 loads.
__device__ __forceinline__ v16h load_bT(const _Float16* __restrict__ w,
                                        int ldk, int lane) {
  const int col = lane & 15;
  const int kb  = (lane & 16) ? 8 : 0;
  const _Float16* p = w + (size_t)col * ldk + kb;
  const v8h lo = *(const v8h*)p;
  const v8h hi = *(const v8h*)(p + 16);
  return concat8(lo, hi);
}

__device__ __forceinline__ v8f wmma_f16(v16h a, v16h b, v8f c) {
  return __builtin_amdgcn_wmma_f32_16x16x32_f16(false, a, false, b,
                                                (short)0, c, false, false);
}

__device__ __forceinline__ float rmax16(float v) {
  v = fmaxf(v, __shfl_xor(v, 1, 32));
  v = fmaxf(v, __shfl_xor(v, 2, 32));
  v = fmaxf(v, __shfl_xor(v, 4, 32));
  v = fmaxf(v, __shfl_xor(v, 8, 32));
  return v;
}
__device__ __forceinline__ float rsum16(float v) {
  v += __shfl_xor(v, 1, 32);
  v += __shfl_xor(v, 2, 32);
  v += __shfl_xor(v, 4, 32);
  v += __shfl_xor(v, 8, 32);
  return v;
}

// ---------------------------------------------------------------------------
// Tensor Data Mover: 2-D tile load Global -> LDS (CDNA5 ISA
// §8; D# group0/group1 bit layout). data_size = 2 bytes (f16), dims in
// elements, tile == tensor extent (no OOB in our use); groups 2/3 + extra
// group zero (2-D tensor).
// ---------------------------------------------------------------------------
__device__ __forceinline__ unsigned int lds_off32(const void* p) {
  // Generic pointers to LDS carry the LDS byte offset in the low 32 bits
  // (aperture addressing, CDNA5 ISA).
  return (unsigned int)(uintptr_t)p;
}

__device__ __forceinline__ void tdm_load_2d_f16(unsigned int lds_addr,
                                                const void* gptr,
                                                unsigned int dim0,
                                                unsigned int dim1,
                                                unsigned int stride0) {
  const unsigned long long ga = (unsigned long long)(uintptr_t)gptr;
  u32x4 g0;
  g0[0] = 1u;                                   // count=1, user D#
  g0[1] = lds_addr;                             // lds_addr  [63:32]
  g0[2] = (unsigned int)ga;                     // global_addr[95:64]
  g0[3] = (unsigned int)(ga >> 32) | (2u << 30);  // ga[120:96] | type=2
  i32x8 g1;
  g1[0] = (int)(1u << 16);                      // data_size code 1 = 2 bytes
  g1[1] = (int)((dim0 & 0xFFFFu) << 16);        // tensor_dim0[15:0] @ 63:48
  g1[2] = (int)(((dim0 >> 16) & 0xFFFFu) | ((dim1 & 0xFFFFu) << 16));
  g1[3] = (int)(((dim1 >> 16) & 0xFFFFu) | ((dim0 & 0xFFFFu) << 16));
                                                // tile_dim0 @ 127:112
  g1[4] = (int)(dim1 & 0xFFFFu);                // tile_dim1; tile_dim2=0
  g1[5] = (int)stride0;                         // tensor_dim0_stride[31:0]
  g1[6] = 0;                                    // stride0[47:32], stride1 lo
  g1[7] = 0;
  const i32x4 z4 = {0, 0, 0, 0};
  const i32x8 z8 = {0, 0, 0, 0, 0, 0, 0, 0};
  // 6-arg toolchain variant: (g0, g1, g2, g3, extra, cpol)
  __builtin_amdgcn_tensor_load_to_lds(g0, g1, z4, z4, z8, 0);
}

// ---------------------------------------------------------------------------
// Kernel 1: fp32 -> f16 convert (grid-stride)
// ---------------------------------------------------------------------------
__global__ void f32_to_f16(const float* __restrict__ in,
                           _Float16* __restrict__ out, int n) {
  for (int i = blockIdx.x * blockDim.x + threadIdx.x; i < n;
       i += gridDim.x * blockDim.x)
    out[i] = (_Float16)in[i];
}

// ---------------------------------------------------------------------------
// Kernel 2/5: C[M x N] = A[M x K](f16) * W[N x K]^T(f16), fp32 accumulate.
// One wave computes a 32x64 output tile: 2 A-frags x 4 B-frags = 8 WMMA per
// 32-deep k-step, with A/B fragment reuse.
// ---------------------------------------------------------------------------
template <bool OUT_HALF>
__global__ void __launch_bounds__(256)
gemm_xwT(const _Float16* __restrict__ A, const _Float16* __restrict__ W,
         void* __restrict__ Cv, int M, int N, int K) {
  const int lane = threadIdx.x & 31;
  const int wave = blockIdx.x * (blockDim.x >> 5) + (threadIdx.x >> 5);
  const int ntn  = N >> 6;  // 64-wide N tiles
  const int ntm  = M >> 5;  // 32-tall M tiles
  if (wave >= ntm * ntn) return;  // wave-uniform; EXEC stays all-ones
  const int m0 = (wave / ntn) << 5;
  const int n0 = (wave % ntn) << 6;

  v8f acc0[4] = {};
  v8f acc1[4] = {};
  for (int k0 = 0; k0 < K; k0 += 32) {
    const v16h a0 = load_a16(A + (size_t)m0 * K + k0, K, lane);
    const v16h a1 = load_a16(A + (size_t)(m0 + 16) * K + k0, K, lane);
    if (k0 + 32 < K) {  // prefetch next A k-slice -> global_prefetch_b8
      __builtin_prefetch(A + (size_t)m0 * K + k0 + 32, 0, 0);
      __builtin_prefetch(A + (size_t)(m0 + 16) * K + k0 + 32, 0, 0);
    }
#pragma unroll
    for (int j = 0; j < 4; ++j) {
      const v16h b = load_bT(W + (size_t)(n0 + 16 * j) * K + k0, K, lane);
      acc0[j] = wmma_f16(a0, b, acc0[j]);
      acc1[j] = wmma_f16(a1, b, acc1[j]);
    }
  }
  const int col = lane & 15;
  const int rb  = (lane & 16) ? 8 : 0;
  if (OUT_HALF) {
    _Float16* C = (_Float16*)Cv;
#pragma unroll
    for (int j = 0; j < 4; ++j)
#pragma unroll
      for (int r = 0; r < 8; ++r) {
        C[(size_t)(m0 + rb + r) * N + n0 + 16 * j + col] = (_Float16)acc0[j][r];
        C[(size_t)(m0 + 16 + rb + r) * N + n0 + 16 * j + col] =
            (_Float16)acc1[j][r];
      }
  } else {
    float* C = (float*)Cv;
#pragma unroll
    for (int j = 0; j < 4; ++j)
#pragma unroll
      for (int r = 0; r < 8; ++r) {
        C[(size_t)(m0 + rb + r) * N + n0 + 16 * j + col] = acc0[j][r];
        C[(size_t)(m0 + 16 + rb + r) * N + n0 + 16 * j + col] = acc1[j][r];
      }
  }
}

// ---------------------------------------------------------------------------
// Kernel 3: RoPE + L2-norm + per-head gain, repack into head-major layouts.
// One wave per (b,h,n) row; lane j owns the even/odd pair (2j, 2j+1).
//   qh  [b,h,n,d] = g[h] * rope(q)/||rope(q)||     (f16)
//   kh  [b,h,n,d] =        rope(k)/||rope(k)||     (f16)
//   vhT [b,h,d,n] = v^T                            (f16)
// ---------------------------------------------------------------------------
__global__ void __launch_bounds__(256)
rope_norm_repack(const _Float16* __restrict__ qkv,
                 const int* __restrict__ tok_pos,
                 const float* __restrict__ qk_scale,
                 _Float16* __restrict__ qh, _Float16* __restrict__ kh,
                 _Float16* __restrict__ vhT) {
  const int lane = threadIdx.x & 31;
  const int wg   = blockIdx.x * (blockDim.x >> 5) + (threadIdx.x >> 5);
  const int n = wg % SEQ;
  const int h = (wg / SEQ) % NUM_HEADS;
  const int b = wg / (SEQ * NUM_HEADS);

  const _Float16* row = qkv + (size_t)(b * SEQ + n) * QKV_N;
  const int d0 = 2 * lane, d1 = 2 * lane + 1;

  const float pos = (float)tok_pos[n];
  // theta^(-2j/64) = exp(-(2j/64)*ln(10000))
  const float inv = __expf(-((float)d0 / 64.0f) * 9.2103403720f);
  float sn, cs;
  __sincosf(pos * inv, &sn, &cs);

  const size_t bh   = (size_t)b * NUM_HEADS + h;
  const size_t base = (bh * SEQ + n) * D_K;

  // ---- Q: rope -> normalize -> * g[h] ----
  float qe = (float)row[h * D_K + d0];
  float qo = (float)row[h * D_K + d1];
  float qre = cs * qe - sn * qo;
  float qro = sn * qe + cs * qo;
  float qs = rsum16(qre * qre + qro * qro);
  qs += __shfl_xor(qs, 16, 32);  // full-wave sum: 32 lanes = one 64-d row
  const float qn = qk_scale[h] / (sqrtf(qs) + 1e-8f);
  qh[base + d0] = (_Float16)(qre * qn);
  qh[base + d1] = (_Float16)(qro * qn);

  // ---- K: rope -> normalize (row-major store) ----
  float ke = (float)row[D_MODEL + h * D_K + d0];
  float ko = (float)row[D_MODEL + h * D_K + d1];
  float kre = cs * ke - sn * ko;
  float kro = sn * ke + cs * ko;
  float ks = rsum16(kre * kre + kro * kro);
  ks += __shfl_xor(ks, 16, 32);
  const float kn = 1.0f / (sqrtf(ks) + 1e-8f);
  kh[base + d0] = (_Float16)(kre * kn);
  kh[base + d1] = (_Float16)(kro * kn);

  // ---- V: transposed store [b,h,d,n] ----
  const size_t vtb = bh * (size_t)D_K * SEQ;
  vhT[vtb + (size_t)d0 * SEQ + n] = row[2 * D_MODEL + h * D_K + d0];
  vhT[vtb + (size_t)d1 * SEQ + n] = row[2 * D_MODEL + h * D_K + d1];
}

// ---------------------------------------------------------------------------
// Kernel 4: flash attention. One wave per (b,h, 16-row query tile).
// K/V tiles streamed Global->LDS by the Tensor Data Mover with ping-pong
// double buffering; S and P*V on WMMA; online softmax in fp32; P staged
// through LDS to reshape C-frag -> A-frag. Output f16 [b,n,h*64+d].
// ---------------------------------------------------------------------------
__global__ void __launch_bounds__(64)
flash_attn(const _Float16* __restrict__ qh, const _Float16* __restrict__ kh,
           const _Float16* __restrict__ vhT, _Float16* __restrict__ oh) {
  // Per-wave: 2 x (32x64) K buffers, 2 x (64x32) V^T buffers, 16x32 P tile.
  __shared__ __align__(16) _Float16 kls[2][2][32 * 64];
  __shared__ __align__(16) _Float16 vls[2][2][64 * 32];
  __shared__ __align__(16) _Float16 plds[2][16 * 32];

  const int lane = threadIdx.x & 31;
  const int w    = threadIdx.x >> 5;
  const int wg   = blockIdx.x * 2 + w;
  const int qt = wg & 127;          // SEQ/16 = 128 query tiles
  const int h  = (wg >> 7) & 15;
  const int b  = wg >> 11;
  const int q0 = qt << 4;

  const size_t bh = (size_t)b * NUM_HEADS + h;
  const _Float16* qb  = qh + (bh * SEQ + q0) * D_K;
  const _Float16* khb = kh + bh * (size_t)SEQ * D_K;   // [key][d]
  const _Float16* vtb = vhT + bh * (size_t)D_K * SEQ;  // [d][key]

  // Q tile A-fragments, loaded once (contraction dim 64 = 2 chunks of 32)
  const v16h aq0 = load_a16(qb, D_K, lane);
  const v16h aq1 = load_a16(qb + 32, D_K, lane);

  const int col = lane & 15;
  const int rb  = (lane & 16) ? 8 : 0;
  const float NEG = -__builtin_inff();

  float mrun[8], lrun[8];
#pragma unroll
  for (int r = 0; r < 8; ++r) { mrun[r] = NEG; lrun[r] = 0.0f; }
  v8f o[4] = {};  // 16 rows x 64 dims accumulator

  const int kbend = q0 + 16;
  // Prologue: TDM-stage key block 0 into buffer 0.
  tdm_load_2d_f16(lds_off32(&kls[w][0][0]), khb, 64, 32, 64);
  tdm_load_2d_f16(lds_off32(&vls[w][0][0]), vtb, 32, 64, SEQ);

  int buf = 0;
  for (int kb = 0; kb < kbend; kb += 32) {
    const int nkb = kb + 32;
    if (nkb < kbend) {
      // WAR: last iteration's ds reads of buf^1 must drain before TDM writes.
      asm volatile("s_wait_dscnt 0" ::: "memory");
      tdm_load_2d_f16(lds_off32(&kls[w][buf ^ 1][0]),
                      khb + (size_t)nkb * D_K, 64, 32, 64);
      tdm_load_2d_f16(lds_off32(&vls[w][buf ^ 1][0]), vtb + nkb, 32, 64, SEQ);
      // Tensor ops retire in order: <=2 outstanding => current block landed.
      asm volatile("s_wait_tensorcnt 0x2" ::: "memory");
    } else {
      asm volatile("s_wait_tensorcnt 0x0" ::: "memory");
    }
    const _Float16* kt = &kls[w][buf][0];  // [key(32)][d(64)]
    const _Float16* vt = &vls[w][buf][0];  // [d(64)][key(32)]

    // ---- S = Q * K^T for 32 keys (two 16-wide C fragments) from LDS ----
    v8f s0 = {}, s1 = {};
    s0 = wmma_f16(aq0, load_bT(kt, D_K, lane), s0);
    s0 = wmma_f16(aq1, load_bT(kt + 32, D_K, lane), s0);
    s1 = wmma_f16(aq0, load_bT(kt + 16 * D_K, D_K, lane), s1);
    s1 = wmma_f16(aq1, load_bT(kt + 16 * D_K + 32, D_K, lane), s1);

    // ---- causal mask + online softmax (lanes<16 own rows 0..7) ----
#pragma unroll
    for (int r = 0; r < 8; ++r) {
      const int qrow = q0 + rb + r;
      const float v0 = (kb + col      <= qrow) ? s0[r] : NEG;
      const float v1 = (kb + 16 + col <= qrow) ? s1[r] : NEG;
      const float mx = rmax16(fmaxf(v0, v1));
      const float mn = fmaxf(mrun[r], mx);
      const float sc = __expf(mrun[r] - mn);  // exp(-inf)=0 on first block
      const float p0 = __expf(v0 - mn);
      const float p1 = __expf(v1 - mn);
      lrun[r] = lrun[r] * sc + rsum16(p0 + p1);
      mrun[r] = mn;
#pragma unroll
      for (int j = 0; j < 4; ++j) o[j][r] *= sc;
      plds[w][(rb + r) * 32 + col]      = (_Float16)p0;
      plds[w][(rb + r) * 32 + 16 + col] = (_Float16)p1;
    }
    // Same-wave LDS RAW on the P tile.
    asm volatile("s_wait_dscnt 0" ::: "memory");

    // ---- O += P * V.  P 16x32 A-frag from LDS; V^T B-frags from LDS ----
    const v16h ap = load_a16(&plds[w][0], 32, lane);
#pragma unroll
    for (int j = 0; j < 4; ++j) {
      const v16h bv = load_bT(vt + (16 * j) * 32, 32, lane);
      o[j] = wmma_f16(ap, bv, o[j]);
    }
    buf ^= 1;
  }

  // ---- normalize and store to [b, n, h*64 + d] (f16) ----
#pragma unroll
  for (int r = 0; r < 8; ++r) {
    const float inv = 1.0f / lrun[r];
    const size_t orow = ((size_t)b * SEQ + q0 + rb + r) * D_MODEL + h * D_K;
#pragma unroll
    for (int j = 0; j < 4; ++j)
      oh[orow + 16 * j + col] = (_Float16)(o[j][r] * inv);
  }
}

// ---------------------------------------------------------------------------
// Host launcher
// ---------------------------------------------------------------------------
extern "C" void kernel_launch(void* const* d_in, const int* in_sizes, int n_in,
                              void* d_out, int out_size, void* d_ws,
                              size_t ws_size, hipStream_t stream) {
  (void)in_sizes; (void)n_in; (void)out_size; (void)ws_size;
  const float* x    = (const float*)d_in[0];
  const int*   pos  = (const int*)d_in[1];
  const float* wqkv = (const float*)d_in[2];
  const float* wo   = (const float*)d_in[3];
  const float* gsc  = (const float*)d_in[4];

  _Float16* ws = (_Float16*)d_ws;
  size_t off = 0;
  _Float16* xh   = ws + off; off += (size_t)ROWS * D_MODEL;      // 4M halves
  _Float16* wqh  = ws + off; off += (size_t)QKV_N * D_MODEL;     // 3M
  _Float16* woh  = ws + off; off += (size_t)D_MODEL * D_MODEL;   // 1M
  _Float16* qkvh = ws + off; off += (size_t)ROWS * QKV_N;        // 12M
  const size_t hsz = (size_t)BATCH * NUM_HEADS * SEQ * D_K;      // 4M each
  _Float16* qh  = ws + off; off += hsz;
  _Float16* kh  = ws + off; off += hsz;
  _Float16* vhT = ws + off; off += hsz;
  _Float16* oh  = ws + off; off += hsz;

  // 1) precision convert
  f32_to_f16<<<2048, 256, 0, stream>>>(x, xh, ROWS * D_MODEL);
  f32_to_f16<<<2048, 256, 0, stream>>>(wqkv, wqh, QKV_N * D_MODEL);
  f32_to_f16<<<1024, 256, 0, stream>>>(wo, woh, D_MODEL * D_MODEL);

  // 2) QKV projection: waves = (4096/32)*(3072/64) = 6144, 8 waves/block
  gemm_xwT<true><<<768, 256, 0, stream>>>(xh, wqh, qkvh, ROWS, QKV_N, D_MODEL);

  // 3) RoPE + QK-norm + gain + head-major repack: 65536 waves, 8/block
  rope_norm_repack<<<8192, 256, 0, stream>>>(qkvh, pos, gsc, qh, kh, vhT);

  // 4) causal flash attention: 2*16*128 = 4096 waves, 2 waves/block (TDM
  //    double buffering uses 34KB LDS per block)
  flash_attn<<<2048, 64, 0, stream>>>(qh, kh, vhT, oh);

  // 5) output projection: waves = 128*16 = 2048, 8 waves/block
  gemm_xwT<false><<<256, 256, 0, stream>>>(oh, woh, d_out, ROWS, D_MODEL,
                                           D_MODEL);
}
